// FeGAN_17858474016783
// MI455X (gfx1250) — matrix operand order repaired and checked by reading, not compiled, not run on hardware
//
#include <hip/hip_runtime.h>
#include <math.h>

#define NNODES 50000
#define NEDGES 800000
#define ETOT   (NEDGES + NNODES)   // 850000 with self loops
#define IN_CH  128
#define HID    64
#define HEADS  8
#define F1     (HEADS * HID)       // 512
#define OUT_CH 4
#define NEG_SLOPE 0.2f
#define EPSV 1e-16f

typedef __attribute__((ext_vector_type(16))) __bf16 v16bf;
typedef __attribute__((ext_vector_type(8)))  __bf16 v8bf;
typedef __attribute__((ext_vector_type(8)))  float  v8f;

// ---------- helpers ----------
__device__ __forceinline__ __bf16 f2bf(float f) {
  unsigned u = __float_as_uint(f);
  u += 0x7FFFu + ((u >> 16) & 1u);          // round-to-nearest-even
  unsigned short s = (unsigned short)(u >> 16);
  return *reinterpret_cast<__bf16*>(&s);
}

__device__ __forceinline__ void get_edge(const int* __restrict__ ei, int e, int& s, int& d) {
  if (e < NEDGES) { s = ei[e]; d = ei[NEDGES + e]; }
  else            { s = e - NEDGES; d = s; }           // self loop
}

__device__ __forceinline__ float atomicMaxF(float* addr, float v) {
  if (v >= 0.0f)
    return __int_as_float(atomicMax((int*)addr, __float_as_int(v)));
  return __uint_as_float(atomicMin((unsigned int*)addr, __float_as_uint(v)));
}

__device__ __forceinline__ float lrelu(float x) { return x > 0.0f ? x : NEG_SLOPE * x; }

// ---------- utility kernels ----------
__global__ void k_f32_to_bf16(const float* __restrict__ in, __bf16* __restrict__ out, int n) {
  int i = blockIdx.x * blockDim.x + threadIdx.x;
  int stride = gridDim.x * blockDim.x;
  for (; i < n; i += stride) out[i] = f2bf(in[i]);
}

__global__ void k_fill(float* __restrict__ p, float v, int n) {
  int i = blockIdx.x * blockDim.x + threadIdx.x;
  int stride = gridDim.x * blockDim.x;
  for (; i < n; i += stride) p[i] = v;
}

// ---------- GEMM1: h1 = x @ W1^T via bf16 WMMA (f32 accum) ----------
// A = x_bf16 [NNODES,128] row-major, B = W1^T with W1 [512,128] row-major.
// One wave computes a 16(M) x 64(N) strip: 4 accum tiles, K-loop of 4 x k32.
// B fragments preloaded into distinct registers per k-step; A loads pipelined
// one k-step ahead so WMMA issue overlaps vmem.
__global__ void __launch_bounds__(128)
k_gemm1_wmma(const __bf16* __restrict__ A, const __bf16* __restrict__ W,
             float* __restrict__ H1) {
  const int lane = threadIdx.x & 31;
  const int wave = threadIdx.x >> 5;
  const int m0 = (blockIdx.x * 4 + wave) * 16;
  if (m0 >= NNODES) return;                 // wave-uniform exit: EXEC all-ones at WMMA
  const int n0   = blockIdx.y * 64;
  const int half = lane >> 4;               // 0: lanes 0-15, 1: lanes 16-31
  const int l16  = lane & 15;

  const __bf16* arow = A + (size_t)(m0 + l16) * IN_CH + half * 16;
  const __bf16* wbase = W + half * 16;

  // A fragment loader: lane half h, row m0+l16 holds K = kb+{h*16..+15}
  auto load_a = [&](int kb) {
    v8bf alo = *(const v8bf*)(arow + kb);
    v8bf ahi = *(const v8bf*)(arow + kb + 8);
    v16bf a;
#pragma unroll
    for (int i = 0; i < 8; ++i) { a[i] = alo[i]; a[i + 8] = ahi[i]; }
    return a;
  };

  v8f acc[4] = {v8f{}, v8f{}, v8f{}, v8f{}};
  v16bf afrag = load_a(0);

#pragma unroll
  for (int ks = 0; ks < 4; ++ks) {
    const int kb = ks * 32;
    // preload all 4 B fragments (distinct registers -> staggered loadcnt waits)
    v16bf bfrag[4];
#pragma unroll
    for (int t = 0; t < 4; ++t) {
      const int col = n0 + t * 16 + l16;    // B 32x16: lane = column, half = K-half
      bfrag[t] = *(const v16bf*)(wbase + (size_t)col * IN_CH + kb);
    }
    // prefetch next k-step's A fragment before the WMMA chain
    v16bf anext;
    if (ks < 3) anext = load_a(kb + 32);
#pragma unroll
    for (int t = 0; t < 4; ++t)
      acc[t] = __builtin_amdgcn_wmma_f32_16x16x32_bf16(
          false, afrag, false, bfrag[t], (short)0, acc[t], false, false);
    if (ks < 3) afrag = anext;
  }

  // D layout: VGPR j -> M = m0 + j + 8*half, N = n0 + t*16 + l16
#pragma unroll
  for (int t = 0; t < 4; ++t) {
    const int n = n0 + t * 16 + l16;
#pragma unroll
    for (int j = 0; j < 8; ++j)
      H1[(size_t)(m0 + half * 8 + j) * F1 + n] = acc[t][j];
  }
}

// ---------- attention logits layer 1: al_src/al_dst [N,8] ----------
__global__ void __launch_bounds__(256)
k_attn_logits1(const float* __restrict__ H1, const float* __restrict__ a_src,
               const float* __restrict__ a_dst, float* __restrict__ als,
               float* __restrict__ ald) {
  const int lane = threadIdx.x & 31;
  const int gw = blockIdx.x * (blockDim.x >> 5) + (threadIdx.x >> 5);
  const int n = gw >> 3, h = gw & 7;
  if (n >= NNODES) return;
  const float* row = H1 + (size_t)n * F1 + h * HID;
  float s0 = 0.f, s1 = 0.f;
#pragma unroll
  for (int i = 0; i < 2; ++i) {
    int c = lane + i * 32;
    float v = row[c];
    s0 += v * a_src[h * HID + c];
    s1 += v * a_dst[h * HID + c];
  }
  for (int off = 16; off; off >>= 1) { s0 += __shfl_xor(s0, off); s1 += __shfl_xor(s1, off); }
  if (lane == 0) { als[n * 8 + h] = s0; ald[n * 8 + h] = s1; }
}

// ---------- layer-1 segment softmax ----------
__global__ void k_edge_max1(const int* __restrict__ ei, const float* __restrict__ als,
                            const float* __restrict__ ald, float* __restrict__ m1) {
  int t = blockIdx.x * blockDim.x + threadIdx.x;
  if (t >= ETOT * 8) return;
  int e = t >> 3, h = t & 7, s, d;
  get_edge(ei, e, s, d);
  atomicMaxF(&m1[d * 8 + h], lrelu(als[s * 8 + h] + ald[d * 8 + h]));
}

__global__ void k_edge_exp1(const int* __restrict__ ei, const float* __restrict__ als,
                            const float* __restrict__ ald, const float* __restrict__ m1,
                            float* __restrict__ ee1, float* __restrict__ den1) {
  int t = blockIdx.x * blockDim.x + threadIdx.x;
  if (t >= ETOT * 8) return;
  int e = t >> 3, h = t & 7, s, d;
  get_edge(ei, e, s, d);
  float x = lrelu(als[s * 8 + h] + ald[d * 8 + h]);
  float ex = __expf(x - m1[d * 8 + h]);
  ee1[(size_t)e * 8 + h] = ex;
  atomicAdd(&den1[d * 8 + h], ex);
}

// one wave per edge: out1[dst] += alpha_h * h1[src] (512 channels)
__global__ void __launch_bounds__(256)
k_edge_aggr1(const int* __restrict__ ei, const float* __restrict__ H1,
             const float* __restrict__ ee1, const float* __restrict__ den1,
             float* __restrict__ out1) {
  const int lane = threadIdx.x & 31;
  const int e = blockIdx.x * (blockDim.x >> 5) + (threadIdx.x >> 5);
  if (e >= ETOT) return;
  int s, d;
  get_edge(ei, e, s, d);
  const float* hsrc = H1 + (size_t)s * F1;
  float* od = out1 + (size_t)d * F1;
#pragma unroll
  for (int h = 0; h < HEADS; ++h) {
    float alpha = ee1[(size_t)e * 8 + h] / (den1[d * 8 + h] + EPSV);
    int c = h * HID + lane;
    atomicAdd(&od[c],      alpha * hsrc[c]);
    atomicAdd(&od[c + 32], alpha * hsrc[c + 32]);
  }
}

// ---------- ELU(out1 + b1) in place ----------
__global__ void k_elu_bias(float* __restrict__ z, const float* __restrict__ b1) {
  size_t i = (size_t)blockIdx.x * blockDim.x + threadIdx.x;
  size_t stride = (size_t)gridDim.x * blockDim.x;
  const size_t total = (size_t)NNODES * F1;
  for (; i < total; i += stride) {
    float v = z[i] + b1[i & (F1 - 1)];
    z[i] = v > 0.f ? v : (__expf(v) - 1.f);
  }
}

// ---------- layer 2 linear (512 -> 4) + attention logits, wave per node ----------
__global__ void __launch_bounds__(256)
k_lin2(const float* __restrict__ Z, const float* __restrict__ W2,
       const float* __restrict__ a2s, const float* __restrict__ a2d,
       float* __restrict__ H2, float* __restrict__ als2, float* __restrict__ ald2) {
  const int lane = threadIdx.x & 31;
  const int n = blockIdx.x * (blockDim.x >> 5) + (threadIdx.x >> 5);
  if (n >= NNODES) return;
  const float* z = Z + (size_t)n * F1;
  float acc[4] = {0.f, 0.f, 0.f, 0.f};
  for (int k = lane; k < F1; k += 32) {
    float zv = z[k];
#pragma unroll
    for (int j = 0; j < 4; ++j) acc[j] += zv * W2[j * F1 + k];
  }
#pragma unroll
  for (int j = 0; j < 4; ++j)
    for (int off = 16; off; off >>= 1) acc[j] += __shfl_xor(acc[j], off);
  if (lane == 0) {
    float s0 = 0.f, s1 = 0.f;
#pragma unroll
    for (int j = 0; j < 4; ++j) {
      H2[n * 4 + j] = acc[j];
      s0 += acc[j] * a2s[j];
      s1 += acc[j] * a2d[j];
    }
    als2[n] = s0;
    ald2[n] = s1;
  }
}

// ---------- layer-2 segment softmax + aggregation (thread per edge) ----------
__global__ void k_edge_max2(const int* __restrict__ ei, const float* __restrict__ als2,
                            const float* __restrict__ ald2, float* __restrict__ m2) {
  int e = blockIdx.x * blockDim.x + threadIdx.x;
  if (e >= ETOT) return;
  int s, d;
  get_edge(ei, e, s, d);
  atomicMaxF(&m2[d], lrelu(als2[s] + ald2[d]));
}

__global__ void k_edge_exp2(const int* __restrict__ ei, const float* __restrict__ als2,
                            const float* __restrict__ ald2, const float* __restrict__ m2,
                            float* __restrict__ ee2, float* __restrict__ den2) {
  int e = blockIdx.x * blockDim.x + threadIdx.x;
  if (e >= ETOT) return;
  int s, d;
  get_edge(ei, e, s, d);
  float ex = __expf(lrelu(als2[s] + ald2[d]) - m2[d]);
  ee2[e] = ex;
  atomicAdd(&den2[d], ex);
}

__global__ void k_edge_aggr2(const int* __restrict__ ei, const float* __restrict__ H2,
                             const float* __restrict__ ee2, const float* __restrict__ den2,
                             float* __restrict__ out2) {
  int e = blockIdx.x * blockDim.x + threadIdx.x;
  if (e >= ETOT) return;
  int s, d;
  get_edge(ei, e, s, d);
  float alpha = ee2[e] / (den2[d] + EPSV);
#pragma unroll
  for (int j = 0; j < 4; ++j) atomicAdd(&out2[d * 4 + j], alpha * H2[s * 4 + j]);
}

// ---------- bias + log_softmax ----------
__global__ void k_logsoftmax(const float* __restrict__ out2, const float* __restrict__ b2,
                             float* __restrict__ y) {
  int n = blockIdx.x * blockDim.x + threadIdx.x;
  if (n >= NNODES) return;
  float v[4], mx = -INFINITY;
#pragma unroll
  for (int j = 0; j < 4; ++j) { v[j] = out2[n * 4 + j] + b2[j]; mx = fmaxf(mx, v[j]); }
  float s = 0.f;
#pragma unroll
  for (int j = 0; j < 4; ++j) s += __expf(v[j] - mx);
  float lse = mx + __logf(s);
#pragma unroll
  for (int j = 0; j < 4; ++j) y[n * 4 + j] = v[j] - lse;
}

// ---------------------------------------------------------------------------
extern "C" void kernel_launch(void* const* d_in, const int* in_sizes, int n_in,
                              void* d_out, int out_size, void* d_ws, size_t ws_size,
                              hipStream_t stream) {
  const float* x      = (const float*)d_in[0];
  const int*   ei     = (const int*)d_in[1];
  const float* W1     = (const float*)d_in[2];
  const float* a1_src = (const float*)d_in[3];
  const float* a1_dst = (const float*)d_in[4];
  const float* b1     = (const float*)d_in[5];
  const float* W2     = (const float*)d_in[6];
  const float* a2_src = (const float*)d_in[7];
  const float* a2_dst = (const float*)d_in[8];
  const float* b2     = (const float*)d_in[9];
  float* y = (float*)d_out;

  // workspace bump allocator (256B aligned)
  char* p = (char*)d_ws;
  auto take = [&](size_t bytes) {
    char* r = p;
    p += (bytes + 255) & ~(size_t)255;
    return (void*)r;
  };
  __bf16* xbf  = (__bf16*)take((size_t)NNODES * IN_CH * 2);
  __bf16* w1bf = (__bf16*)take((size_t)F1 * IN_CH * 2);
  float* H1    = (float*)take((size_t)NNODES * F1 * 4);
  float* als   = (float*)take((size_t)NNODES * 8 * 4);
  float* ald   = (float*)take((size_t)NNODES * 8 * 4);
  float* m1    = (float*)take((size_t)NNODES * 8 * 4);
  float* den1  = (float*)take((size_t)NNODES * 8 * 4);
  float* ee1   = (float*)take((size_t)ETOT * 8 * 4);
  float* out1  = (float*)take((size_t)NNODES * F1 * 4);  // reused as z1 (ELU in place)
  float* H2    = (float*)take((size_t)NNODES * 4 * 4);
  float* als2  = (float*)take((size_t)NNODES * 4);
  float* ald2  = (float*)take((size_t)NNODES * 4);
  float* m2    = (float*)take((size_t)NNODES * 4);
  float* den2  = (float*)take((size_t)NNODES * 4);
  float* ee2   = (float*)take((size_t)ETOT * 4);
  float* out2  = (float*)take((size_t)NNODES * 4 * 4);

  const float NEG_INF = -INFINITY;

  // conversions
  k_f32_to_bf16<<<(NNODES * IN_CH + 255) / 256, 256, 0, stream>>>(x, xbf, NNODES * IN_CH);
  k_f32_to_bf16<<<(F1 * IN_CH + 255) / 256, 256, 0, stream>>>(W1, w1bf, F1 * IN_CH);

  // init accumulators (every call: replay-safe)
  k_fill<<<(NNODES * 8 + 255) / 256, 256, 0, stream>>>(m1, NEG_INF, NNODES * 8);
  k_fill<<<(NNODES * 8 + 255) / 256, 256, 0, stream>>>(den1, 0.f, NNODES * 8);
  k_fill<<<(NNODES * F1 + 255) / 256, 256, 0, stream>>>(out1, 0.f, NNODES * F1);
  k_fill<<<(NNODES + 255) / 256, 256, 0, stream>>>(m2, NEG_INF, NNODES);
  k_fill<<<(NNODES + 255) / 256, 256, 0, stream>>>(den2, 0.f, NNODES);
  k_fill<<<(NNODES * 4 + 255) / 256, 256, 0, stream>>>(out2, 0.f, NNODES * 4);

  // layer 1
  dim3 g1((3125 + 3) / 4, F1 / 64);  // 782 x 8, 4 waves/block, 16x64 per wave
  k_gemm1_wmma<<<g1, 128, 0, stream>>>(xbf, w1bf, H1);
  k_attn_logits1<<<NNODES, 256, 0, stream>>>(H1, a1_src, a1_dst, als, ald);
  k_edge_max1<<<(ETOT * 8 + 255) / 256, 256, 0, stream>>>(ei, als, ald, m1);
  k_edge_exp1<<<(ETOT * 8 + 255) / 256, 256, 0, stream>>>(ei, als, ald, m1, ee1, den1);
  k_edge_aggr1<<<(ETOT + 7) / 8, 256, 0, stream>>>(ei, H1, ee1, den1, out1);
  k_elu_bias<<<100000, 256, 0, stream>>>(out1, b1);

  // layer 2
  k_lin2<<<(NNODES + 7) / 8, 256, 0, stream>>>(out1, W2, a2_src, a2_dst, H2, als2, ald2);
  k_edge_max2<<<(ETOT + 255) / 256, 256, 0, stream>>>(ei, als2, ald2, m2);
  k_edge_exp2<<<(ETOT + 255) / 256, 256, 0, stream>>>(ei, als2, ald2, m2, ee2, den2);
  k_edge_aggr2<<<(ETOT + 255) / 256, 256, 0, stream>>>(ei, H2, ee2, den2, out2);
  k_logsoftmax<<<(NNODES + 255) / 256, 256, 0, stream>>>(out2, b2, y);
}
